// TemporalDiffAttn_42116449305216
// MI455X (gfx1250) — matrix operand order repaired and checked by reading, not compiled, not run on hardware
//
#include <hip/hip_runtime.h>
#include <hip/hip_bf16.h>

// ---------- types ----------
typedef __attribute__((ext_vector_type(16))) __bf16          v16bf;
typedef __attribute__((ext_vector_type(8)))  float           v8f;
typedef __attribute__((ext_vector_type(16))) unsigned short  u16x16;
typedef __attribute__((ext_vector_type(8)))  unsigned short  u16x8;
typedef __attribute__((ext_vector_type(4)))  float           f32x4;
typedef __attribute__((ext_vector_type(4)))  unsigned int    u32x4;
typedef __attribute__((ext_vector_type(8)))  int             i32x8;
typedef __attribute__((ext_vector_type(4)))  int             i32x4;

#define T_SEQ   2048
#define NHEAD   4
#define HD      32
#define BATCH   8
#define SCALE   0.17677669529663687f   // 1/sqrt(32)
#define LAMB0   0.8f

// float -> bf16 bits, round-to-nearest-even
__device__ __forceinline__ unsigned short f2bf(float f) {
  unsigned int x = __float_as_uint(f);
  x += 0x7FFFu + ((x >> 16) & 1u);
  return (unsigned short)(x >> 16);
}
__device__ __forceinline__ v16bf ld_frag32(const unsigned short* p) {
  u16x16 r = *(const u16x16*)p;            // 32B contiguous
  return __builtin_bit_cast(v16bf, r);
}
__device__ __forceinline__ v16bf combine8(u16x8 lo, u16x8 hi) {
  u16x16 t;
#pragma unroll
  for (int i = 0; i < 8; ++i) { t[i] = lo[i]; t[i + 8] = hi[i]; }
  return __builtin_bit_cast(v16bf, t);
}
__device__ __forceinline__ v8f wmma_bf16(v16bf a, v16bf b, v8f c) {
  return __builtin_amdgcn_wmma_f32_16x16x32_bf16(false, a, false, b, (short)0, c,
                                                 false, false);
}

// ---- Tensor Data Mover: 2D tile (bf16) global -> LDS  (ISA 8.3/8.4) ----
// clang-23 / therock-10.0 builtin form: (v4u, v8i, v4i, v4i, v8i, i32 cpol)
__device__ __forceinline__ void tdm_load_2d(unsigned int lds_off, const void* gptr,
                                            unsigned int tensor_d0, unsigned int tensor_d1,
                                            unsigned int tile_d0, unsigned int tile_d1,
                                            unsigned long long stride0) {
  unsigned long long ga = (unsigned long long)(uintptr_t)gptr;
  u32x4 g0;
  g0[0] = 1u;                                                    // count=1, no gather
  g0[1] = lds_off;                                               // lds_addr
  g0[2] = (unsigned int)ga;                                      // global_addr lo
  g0[3] = ((unsigned int)(ga >> 32) & 0x01FFFFFFu) | (2u << 30); // addr hi | type=2
  i32x8 g1;
  g1[0] = (int)(1u << 16);                                       // data_size = 2B
  g1[1] = (int)((tensor_d0 & 0xFFFFu) << 16);                    // tensor_dim0[15:0]
  g1[2] = (int)((tensor_d0 >> 16) | ((tensor_d1 & 0xFFFFu) << 16));
  g1[3] = (int)((tensor_d1 >> 16) | ((tile_d0 & 0xFFFFu) << 16));
  g1[4] = (int)(tile_d1 & 0xFFFFu);                              // tile_dim1, tile_dim2=0
  g1[5] = (int)(unsigned int)(stride0 & 0xFFFFFFFFu);            // tensor_dim0_stride lo
  g1[6] = (int)((unsigned int)(stride0 >> 32) & 0xFFFFu);        // stride hi, dim1_stride=0
  g1[7] = 0;
  i32x4 z4 = {0, 0, 0, 0};
  i32x8 z8 = {0, 0, 0, 0, 0, 0, 0, 0};
  __builtin_amdgcn_tensor_load_to_lds(g0, g1, z4, z4, z8, 0);
}

// ============================================================
// Kernel 1: layernorm + fused QKV projection, writes bf16 staging
//   qb,kb: [B*H][T][64] row-major ;  vtb: [B*H][64][T] (dim-major)
// ============================================================
__global__ void ln_qkv_kernel(const float* __restrict__ tokens,
                              const float* __restrict__ ln_w,
                              const float* __restrict__ ln_b,
                              const float* __restrict__ wq,
                              const float* __restrict__ wk,
                              const float* __restrict__ wv,
                              unsigned short* __restrict__ qb,
                              unsigned short* __restrict__ kb,
                              unsigned short* __restrict__ vtb) {
  const int t   = blockIdx.x;        // flat token 0..B*T-1
  const int tid = threadIdx.x;       // 0..255
  __shared__ float xs[128];
  __shared__ float red[2];

  if (tid < 128) xs[tid] = tokens[(size_t)t * 128 + tid];
  __syncthreads();
  if (tid < 32) {
    float s = 0.f, s2 = 0.f;
    for (int i = tid; i < 128; i += 32) { float v = xs[i]; s += v; s2 += v * v; }
#pragma unroll
    for (int m = 16; m; m >>= 1) { s += __shfl_xor(s, m, 32); s2 += __shfl_xor(s2, m, 32); }
    if (tid == 0) {
      float mu  = s * (1.f / 128.f);
      float var = s2 * (1.f / 128.f) - mu * mu;
      red[0] = mu; red[1] = rsqrtf(var + 1e-5f);
    }
  }
  __syncthreads();
  float mu = red[0], rs = red[1];
  if (tid < 128) xs[tid] = (xs[tid] - mu) * rs * ln_w[tid] + ln_b[tid];
  __syncthreads();

  const int tt = t & (T_SEQ - 1);
  const int b  = t >> 11;
#pragma unroll
  for (int c3 = 0; c3 < 3; ++c3) {
    const float* w = (c3 == 0) ? wq : (c3 == 1) ? wk : wv;
    int col = tid;
    float acc = 0.f;
#pragma unroll 8
    for (int i = 0; i < 128; ++i) acc = fmaf(xs[i], w[i * 256 + col], acc);
    unsigned short bits = f2bf(acc);
    int h = col >> 6, d = col & 63;
    size_t bh = (size_t)(b * NHEAD + h);
    if (c3 == 0)      qb [(bh * T_SEQ + tt) * 64 + d] = bits;
    else if (c3 == 1) kb [(bh * T_SEQ + tt) * 64 + d] = bits;
    else              vtb[(bh * 64 + d) * T_SEQ + tt] = bits;
  }
}

// ============================================================
// Kernel 2: dual-softmax flash attention
//   TDM-staged K/V chunks (double-buffered LDS), WMMA bf16, f32 accum.
//   grid (B*H, T/128), 8 waves/block, each wave owns 16 queries.
// ============================================================
__global__ __launch_bounds__(256) void attn_kernel(
    const unsigned short* __restrict__ qb,
    const unsigned short* __restrict__ kb,
    const unsigned short* __restrict__ vtb,
    const float* __restrict__ lq1, const float* __restrict__ lk1,
    const float* __restrict__ lq2, const float* __restrict__ lk2,
    const float* __restrict__ sig_s_p, const float* __restrict__ sig_n_p,
    float* __restrict__ O) {
  const int bh   = blockIdx.x;
  const int lane = threadIdx.x & 31;
  const int w    = threadIdx.x >> 5;
  const int qt0  = blockIdx.y * 128 + w * 16;
  const int nlo  = lane & 15;
  const int kh   = lane >> 4;

  __shared__ unsigned short Kbuf[2][32][64];   // [key][dim]   4KB each
  __shared__ unsigned short Vbuf[2][64][32];   // [dim][key]   4KB each
  __shared__ float          Sc[8][16][32];     // wave-private score bounce
  __shared__ float          Brd[8][2][16];     // wave-private row broadcast

  float e1 = 0.f, e2 = 0.f;
#pragma unroll 8
  for (int i = 0; i < HD; ++i) { e1 += lq1[i] * lk1[i]; e2 += lq2[i] * lk2[i]; }
  const float lam = __expf(e1) - __expf(e2) + LAMB0;
  const float ss = fmaxf(sig_s_p[0], 1.f), sn = fmaxf(sig_n_p[0], 1.f);
  const float c1b = 0.5f / (ss * ss), c2b = 0.5f / (sn * sn);

  // Q1/Q2 A-fragments (16x32 bf16)
  v16bf A1, A2;
  {
    const unsigned short* r = qb + ((size_t)bh * T_SEQ + qt0 + nlo) * 64;
    A1 = combine8(*(const u16x8*)(r + kh * 16),      *(const u16x8*)(r + 8 + kh * 16));
    A2 = combine8(*(const u16x8*)(r + 32 + kh * 16), *(const u16x8*)(r + 40 + kh * 16));
  }

  v8f acc1[4] = {}, acc2[4] = {};
  float m1 = -1e30f, l1 = 0.f, m2 = -1e30f, l2 = 0.f;   // row nlo stats

  const unsigned short* kbase = kb  + (size_t)bh * T_SEQ * 64;
  const unsigned short* vbase = vtb + (size_t)bh * 64 * T_SEQ;

  if (w == 0) {   // wave-uniform: EXEC untouched
    tdm_load_2d((unsigned int)(uintptr_t)&Kbuf[0][0][0], kbase, 64, T_SEQ, 64, 32, 64);
    tdm_load_2d((unsigned int)(uintptr_t)&Vbuf[0][0][0], vbase, T_SEQ, 64, 32, 64, T_SEQ);
  }

  for (int j = 0; j < T_SEQ / 32; ++j) {
    const int kt = j * 32, p = j & 1;
    if (w == 0) {
      if (j + 1 < T_SEQ / 32) {
        tdm_load_2d((unsigned int)(uintptr_t)&Kbuf[1 - p][0][0],
                    kbase + (size_t)(kt + 32) * 64, 64, T_SEQ, 64, 32, 64);
        tdm_load_2d((unsigned int)(uintptr_t)&Vbuf[1 - p][0][0],
                    vbase + (kt + 32), T_SEQ, 64, 32, 64, T_SEQ);
        __builtin_amdgcn_s_wait_tensorcnt(2);   // chunk j's pair complete (in-order)
      } else {
        __builtin_amdgcn_s_wait_tensorcnt(0);
      }
    }
    __syncthreads();                            // buf[p] ready for all waves

    const unsigned short* Kb = &Kbuf[p][0][0];
    const unsigned short* Vb = &Vbuf[p][0][0];

    v8f z = {};
    v8f s1a = wmma_bf16(A1, ld_frag32(Kb + (size_t)nlo        * 64 +      kh * 16), z);
    v8f s1b = wmma_bf16(A1, ld_frag32(Kb + (size_t)(16 + nlo) * 64 +      kh * 16), z);
    v8f s2a = wmma_bf16(A2, ld_frag32(Kb + (size_t)nlo        * 64 + 32 + kh * 16), z);
    v8f s2b = wmma_bf16(A2, ld_frag32(Kb + (size_t)(16 + nlo) * 64 + 32 + kh * 16), z);

    float* sc = &Sc[w][0][0];
    const float ka = (float)(kt + nlo);
    v16bf PA1, PA2;

    // ---------- matrix 1 ----------
#pragma unroll
    for (int r = 0; r < 8; ++r) {
      const float q  = (float)(qt0 + r + 8 * kh);
      const float ra = ka - q, rb = ra + 16.f;
      sc[(r + 8 * kh) * 32 + nlo]      = s1a[r] * SCALE - ra * ra * c1b;
      sc[(r + 8 * kh) * 32 + 16 + nlo] = s1b[r] * SCALE - rb * rb * c1b;
    }
    {
      f32x4 c0 = *(const f32x4*)&Sc[w][nlo][kh * 8];
      f32x4 c1 = *(const f32x4*)&Sc[w][nlo][kh * 8 + 4];
      f32x4 c2 = *(const f32x4*)&Sc[w][nlo][16 + kh * 8];
      f32x4 c3 = *(const f32x4*)&Sc[w][nlo][16 + kh * 8 + 4];
      float vv[16];
#pragma unroll
      for (int i = 0; i < 4; ++i) { vv[i] = c0[i]; vv[4+i] = c1[i]; vv[8+i] = c2[i]; vv[12+i] = c3[i]; }
      float mx = vv[0];
#pragma unroll
      for (int i = 1; i < 16; ++i) mx = fmaxf(mx, vv[i]);
      mx = fmaxf(mx, __shfl_xor(mx, 16, 32));
      float mn = fmaxf(m1, mx);
      float corr = __expf(m1 - mn);
      float sm = 0.f;
      u16x16 pk;
#pragma unroll
      for (int i = 0; i < 16; ++i) { float pe = __expf(vv[i] - mn); sm += pe; pk[i] = f2bf(pe); }
      sm += __shfl_xor(sm, 16, 32);
      l1 = l1 * corr + sm; m1 = mn;
      Brd[w][0][nlo] = corr;            // both halves write identical value
      PA1 = __builtin_bit_cast(v16bf, pk);
    }
    // ---------- matrix 2 ----------
#pragma unroll
    for (int r = 0; r < 8; ++r) {
      const float q  = (float)(qt0 + r + 8 * kh);
      const float ra = ka - q, rb = ra + 16.f;
      sc[(r + 8 * kh) * 32 + nlo]      = s2a[r] * SCALE - ra * ra * c2b;
      sc[(r + 8 * kh) * 32 + 16 + nlo] = s2b[r] * SCALE - rb * rb * c2b;
    }
    {
      f32x4 c0 = *(const f32x4*)&Sc[w][nlo][kh * 8];
      f32x4 c1 = *(const f32x4*)&Sc[w][nlo][kh * 8 + 4];
      f32x4 c2 = *(const f32x4*)&Sc[w][nlo][16 + kh * 8];
      f32x4 c3 = *(const f32x4*)&Sc[w][nlo][16 + kh * 8 + 4];
      float vv[16];
#pragma unroll
      for (int i = 0; i < 4; ++i) { vv[i] = c0[i]; vv[4+i] = c1[i]; vv[8+i] = c2[i]; vv[12+i] = c3[i]; }
      float mx = vv[0];
#pragma unroll
      for (int i = 1; i < 16; ++i) mx = fmaxf(mx, vv[i]);
      mx = fmaxf(mx, __shfl_xor(mx, 16, 32));
      float mn = fmaxf(m2, mx);
      float corr = __expf(m2 - mn);
      float sm = 0.f;
      u16x16 pk;
#pragma unroll
      for (int i = 0; i < 16; ++i) { float pe = __expf(vv[i] - mn); sm += pe; pk[i] = f2bf(pe); }
      sm += __shfl_xor(sm, 16, 32);
      l2 = l2 * corr + sm; m2 = mn;
      Brd[w][1][nlo] = corr;
      PA2 = __builtin_bit_cast(v16bf, pk);
    }

    // correction factors for the rows this lane holds in D-fragments
    float cr1[8], cr2[8];
    {
      f32x4 a0 = *(const f32x4*)&Brd[w][0][kh * 8];
      f32x4 a1 = *(const f32x4*)&Brd[w][0][kh * 8 + 4];
      f32x4 b0 = *(const f32x4*)&Brd[w][1][kh * 8];
      f32x4 b1 = *(const f32x4*)&Brd[w][1][kh * 8 + 4];
#pragma unroll
      for (int i = 0; i < 4; ++i) { cr1[i] = a0[i]; cr1[4+i] = a1[i]; cr2[i] = b0[i]; cr2[4+i] = b1[i]; }
    }
#pragma unroll
    for (int g = 0; g < 4; ++g) {
#pragma unroll
      for (int r = 0; r < 8; ++r) { acc1[g][r] *= cr1[r]; acc2[g][r] *= cr2[r]; }
      v16bf Vf = combine8(*(const u16x8*)(Vb + (size_t)(g * 16 + nlo) * 32 + kh * 8), *(const u16x8*)(Vb + (size_t)(g * 16 + nlo) * 32 + 16 + kh * 8));
      acc1[g] = wmma_bf16(PA1, Vf, acc1[g]);
      acc2[g] = wmma_bf16(PA2, Vf, acc2[g]);
    }
    __syncthreads();                            // everyone done reading buf[p]
  }

  // final normalization: out = acc1/l1 - lam*acc2/l2
  Brd[w][0][nlo] = 1.f / l1;
  Brd[w][1][nlo] = lam / l2;
  float i1[8], i2[8];
  {
    f32x4 a0 = *(const f32x4*)&Brd[w][0][kh * 8];
    f32x4 a1 = *(const f32x4*)&Brd[w][0][kh * 8 + 4];
    f32x4 b0 = *(const f32x4*)&Brd[w][1][kh * 8];
    f32x4 b1 = *(const f32x4*)&Brd[w][1][kh * 8 + 4];
#pragma unroll
    for (int i = 0; i < 4; ++i) { i1[i] = a0[i]; i1[4+i] = a1[i]; i2[i] = b0[i]; i2[4+i] = b1[i]; }
  }
#pragma unroll
  for (int r = 0; r < 8; ++r) {
    int row = qt0 + r + 8 * kh;
#pragma unroll
    for (int g = 0; g < 4; ++g) {
      O[((size_t)bh * T_SEQ + row) * 64 + g * 16 + nlo] = acc1[g][r] * i1[r] - acc2[g][r] * i2[r];
    }
  }
}

// ============================================================
// Kernel 3: per-head layernorm, *0.2, @wo, residual add
// ============================================================
__global__ void out_kernel(const float* __restrict__ O,
                           const float* __restrict__ tokens,
                           const float* __restrict__ hn_w,
                           const float* __restrict__ hn_b,
                           const float* __restrict__ wo,
                           float* __restrict__ out) {
  const int t   = blockIdx.x;
  const int tid = threadIdx.x;      // 0..127
  const int b = t >> 11, tt = t & (T_SEQ - 1);
  __shared__ float os[256];
  __shared__ float stats[8];

  for (int i = tid; i < 256; i += 128) {
    int h = i >> 6, d = i & 63;
    os[i] = O[(((size_t)(b * NHEAD + h)) * T_SEQ + tt) * 64 + d];
  }
  __syncthreads();
  if (tid < 4) {
    float s = 0.f, s2 = 0.f;
    for (int d = 0; d < 64; ++d) { float v = os[tid * 64 + d]; s += v; s2 += v * v; }
    float mu  = s * (1.f / 64.f);
    float var = s2 * (1.f / 64.f) - mu * mu;
    stats[tid * 2] = mu; stats[tid * 2 + 1] = rsqrtf(var + 1e-5f);
  }
  __syncthreads();
  for (int i = tid; i < 256; i += 128) {
    int h = i >> 6, d = i & 63;
    os[i] = ((os[i] - stats[h * 2]) * stats[h * 2 + 1] * hn_w[d] + hn_b[d]) * 0.2f;
  }
  __syncthreads();
  float acc = tokens[(size_t)t * 128 + tid];
#pragma unroll 8
  for (int j = 0; j < 256; ++j) acc = fmaf(os[j], wo[j * 128 + tid], acc);
  out[(size_t)t * 128 + tid] = acc;
}

// ============================================================
extern "C" void kernel_launch(void* const* d_in, const int* in_sizes, int n_in,
                              void* d_out, int out_size, void* d_ws, size_t ws_size,
                              hipStream_t stream) {
  const float* tokens = (const float*)d_in[0];
  const float* ln_w   = (const float*)d_in[1];
  const float* ln_b   = (const float*)d_in[2];
  const float* wq     = (const float*)d_in[3];
  const float* wk     = (const float*)d_in[4];
  const float* wv     = (const float*)d_in[5];
  const float* wo     = (const float*)d_in[6];
  const float* lq1    = (const float*)d_in[7];
  const float* lk1    = (const float*)d_in[8];
  const float* lq2    = (const float*)d_in[9];
  const float* lk2    = (const float*)d_in[10];
  const float* sig_s  = (const float*)d_in[11];
  const float* sig_n  = (const float*)d_in[12];
  const float* hn_w   = (const float*)d_in[13];
  const float* hn_b   = (const float*)d_in[14];

  char* ws = (char*)d_ws;
  const size_t BHT64 = (size_t)BATCH * NHEAD * T_SEQ * 64;
  unsigned short* qb  = (unsigned short*)(ws);
  unsigned short* kb  = (unsigned short*)(ws + BHT64 * 2);
  unsigned short* vtb = (unsigned short*)(ws + BHT64 * 4);
  float*          O   = (float*)        (ws + BHT64 * 6);

  float* outp = (float*)d_out;

  ln_qkv_kernel<<<dim3(BATCH * T_SEQ), dim3(256), 0, stream>>>(
      tokens, ln_w, ln_b, wq, wk, wv, qb, kb, vtb);
  attn_kernel<<<dim3(BATCH * NHEAD, T_SEQ / 128), dim3(256), 0, stream>>>(
      qb, kb, vtb, lq1, lk1, lq2, lk2, sig_s, sig_n, O);
  out_kernel<<<dim3(BATCH * T_SEQ), dim3(128), 0, stream>>>(
      O, tokens, hn_w, hn_b, wo, outp);
}
